// Net_52621939311002
// MI455X (gfx1250) — matrix operand order repaired and checked by reading, not compiled, not run on hardware
//
#include <hip/hip_runtime.h>
#include <cmath>

typedef _Float16 half_t;
typedef __attribute__((ext_vector_type(16))) _Float16 v16h;
typedef __attribute__((ext_vector_type(8)))  float    v8f;
typedef __attribute__((ext_vector_type(4)))  unsigned int u32x4;

// LDS tile strides (halves); all multiples of 8 so 16B-aligned b128 loads work.
#define SHS 136          // state tile stride  (K=128 + pad)
#define XS  296          // feature tile stride (N0=288 + pad)
#define H1S 136          // act-hidden tile stride (N1=128 + pad)
#define SH_BYTES (16*SHS*2)            // 4352
#define X_BYTES  (16*XS*2)             // 9472
#define H1_BYTES (16*H1S*2)            // 4352
#define PER_WAVE (SH_BYTES + X_BYTES + H1_BYTES)   // 18176, 16B aligned
#define WAVES 8

// ---------------- WMMA helpers ----------------

__device__ __forceinline__ v8f wmma_f16(v16h a, v16h b, v8f c) {
  // D = A*B + C, f16 inputs, f32 accumulate
  return __builtin_amdgcn_wmma_f32_16x16x32_f16(false, a, false, b, (short)0, c, false, false);
}

// A-matrix fragment (16x32 f16) from row-major LDS tile [16][stride]:
// lane<16: M=lane, halves 0..7 = K k0..k0+7, halves 8..15 = K k0+16..k0+23
// lane>=16: M=lane-16, K ranges shifted by +8.
__device__ __forceinline__ v16h ldsA(const half_t* base, int stride, int k0, int lane) {
  int row = lane & 15;
  int ko  = (lane >> 4) << 3;
  const half_t* p = base + row * stride + k0 + ko;
  union { v16h v; u32x4 u[2]; } t;
  t.u[0] = *(const u32x4*)p;
  t.u[1] = *(const u32x4*)(p + 16);
  return t.v;
}

// B-matrix fragment (32x16 f16) from global weights stored [N][K] (transposed):
// lane%16 = column N; halves 0..7 = K k0+8h..k0+8h+7, halves 8..15 = K k0+16+8h..k0+23+8h (h = lane/16).
__device__ __forceinline__ v16h ldgB(const half_t* __restrict__ WT, int Ks, int n0, int k0, int lane) {
  const half_t* p = WT + (size_t)(n0 + (lane & 15)) * Ks + k0 + ((lane >> 4) << 3);
  union { v16h v; u32x4 u[2]; } t;
  t.u[0] = *(const u32x4*)p;
  t.u[1] = *(const u32x4*)(p + 16);
  return t.v;
}

// ---------------- weight prep ----------------

__device__ __forceinline__ int pidx(int x, int y) {
  int xf = x < 7 - x ? x : 7 - x;
  int yf = y < 7 - y ? y : 7 - y;
  int a = xf < yf ? xf : yf;
  int b = xf < yf ? yf : xf;
  return a * 4 - a * (a - 1) / 2 + (b - a);   // index into the 10-entry Dic
}

__global__ void net_prep(const float* __restrict__ myp, const float* __restrict__ hisp,
                         const float* __restrict__ w1, const float* __restrict__ w2,
                         const float* __restrict__ wv,
                         half_t* __restrict__ WcT,   // [288][128]
                         half_t* __restrict__ W1T,   // [128][288]
                         half_t* __restrict__ W2T,   // [64][128]
                         half_t* __restrict__ WvT)   // [64][288]
{
  const int tid = threadIdx.x;   // single block, 256 threads
  const int TOT = 288*128 + 128*288 + 64*128 + 64*288;  // contiguous in ws
  for (int i = tid; i < TOT; i += 256) WcT[i] = (half_t)0.f;
  __syncthreads();

  // Fold the symmetric conv into WcT: feature f = sum_k state_flat[k] * WcT[f][k]
  for (int t = tid; t < 384; t += 256) {       // 2 planes * 3 channels * 64 cells
    int p = t & 1;
    int j = (t >> 1) % 3;
    int cell = t / 6;
    int x = cell >> 3, y = cell & 7;
    const float* par = p ? hisp : myp;         // [3,10]
    float d = par[j * 10 + pidx(x, y)];
    half_t h = (half_t)d;
    int k = p * 64 + x * 8 + y;                // state flat index [2,8,8]
    int fr = (j * 8 + x) * 2 + p;              // row feature (i=x)
    int fc = 48 + (j * 8 + y) * 2 + p;         // col feature (i=y); D symmetric
    int md = (y >= x) ? (y - x) : (x - y + 7); // main diagonal id
    int fm = 96 + (j * 15 + md) * 2 + p;
    int fa = 186 + (j * 15 + x + y) * 2 + p;   // anti diagonal id = x+y
    WcT[fr * 128 + k] = h;
    WcT[fc * 128 + k] = h;
    WcT[fm * 128 + k] = h;
    WcT[fa * 128 + k] = h;
  }

  for (int i = tid; i < 100 * 276; i += 256) { // w_act1 [100,276] -> [128][288]
    int n = i / 276, k = i % 276;
    W1T[n * 288 + k] = (half_t)w1[i];
  }
  for (int i = tid; i < 64 * 100; i += 256) {  // w_act2 [64,100] -> [64][128]
    int n = i / 100, k = i % 100;
    W2T[n * 128 + k] = (half_t)w2[i];
  }
  for (int i = tid; i < 64 * 276; i += 256) {  // w_val1 [64,276] -> [64][288]
    int n = i / 276, k = i % 276;
    WvT[n * 288 + k] = (half_t)wv[i];
  }
}

// ---------------- fused forward ----------------

__launch_bounds__(256)
__global__ void net_main(const float* __restrict__ state,
                         const float* __restrict__ b1, const float* __restrict__ b2,
                         const float* __restrict__ bv1, const float* __restrict__ wv2,
                         const float* __restrict__ bv2,
                         const half_t* __restrict__ WcT, const half_t* __restrict__ W1T,
                         const half_t* __restrict__ W2T, const half_t* __restrict__ WvT,
                         float* __restrict__ out_act, float* __restrict__ out_val)
{
  extern __shared__ char lds[];
  const int lane = threadIdx.x & 31;
  const int wave = threadIdx.x >> 5;
  const size_t rowBase = (size_t)blockIdx.x * (16 * WAVES) + wave * 16;

  char* wb = lds + wave * PER_WAVE;            // per-wave private region
  half_t* Sh  = (half_t*)wb;                   // [16][SHS] state f16
  half_t* X   = (half_t*)(wb + SH_BYTES);      // [16][XS]  features f16
  half_t* H1  = (half_t*)(wb + SH_BYTES + X_BYTES); // [16][H1S] act hidden f16
  float*  SCR = (float*)wb;                    // [16][64] f32, aliases Sh (dead after GEMM0)
  float*  RM  = SCR + 16 * 64;                 // [16] per-row logsumexp

  const int colL = lane & 15;                  // D-tile column within n-tile
  const int rb   = (lane >> 4) << 3;           // D-tile row base (0 or 8)

  // Stage state tile -> LDS (f32 -> f16), coalesced
  const float* sp = state + rowBase * 128;
  for (int idx = lane; idx < 16 * 128; idx += 32) {
    int r = idx >> 7, k = idx & 127;
    Sh[r * SHS + k] = (half_t)sp[idx];
  }

  // GEMM0: x = relu(S[16x128] * WcT^T) -> X[16x288]
  for (int nt = 0; nt < 18; ++nt) {
    v8f c = {0.f, 0.f, 0.f, 0.f, 0.f, 0.f, 0.f, 0.f};
    #pragma unroll
    for (int kt = 0; kt < 4; ++kt)
      c = wmma_f16(ldsA(Sh, SHS, kt * 32, lane), ldgB(WcT, 128, nt * 16, kt * 32, lane), c);
    int col = nt * 16 + colL;
    #pragma unroll
    for (int r = 0; r < 8; ++r)
      X[(rb + r) * XS + col] = (half_t)fmaxf(c[r], 0.f);
  }

  // GEMM1: h_a = relu(X[16x288] * W1T^T + b1) -> H1[16x128] (cols >=100 stay 0)
  for (int nt = 0; nt < 8; ++nt) {
    v8f c = {0.f, 0.f, 0.f, 0.f, 0.f, 0.f, 0.f, 0.f};
    #pragma unroll
    for (int kt = 0; kt < 9; ++kt)
      c = wmma_f16(ldsA(X, XS, kt * 32, lane), ldgB(W1T, 288, nt * 16, kt * 32, lane), c);
    int col = nt * 16 + colL;
    float bias = (col < 100) ? b1[col] : 0.f;
    #pragma unroll
    for (int r = 0; r < 8; ++r)
      H1[(rb + r) * H1S + col] = (half_t)fmaxf(c[r] + bias, 0.f);
  }

  __syncthreads();   // Sh is dead; fence before reusing its bytes as SCR (f32)

  // GEMM2: logits = H1[16x128] * W2T^T + b2 -> SCR[16x64] f32
  for (int nt = 0; nt < 4; ++nt) {
    v8f c = {0.f, 0.f, 0.f, 0.f, 0.f, 0.f, 0.f, 0.f};
    #pragma unroll
    for (int kt = 0; kt < 4; ++kt)
      c = wmma_f16(ldsA(H1, H1S, kt * 32, lane), ldgB(W2T, 128, nt * 16, kt * 32, lane), c);
    int col = nt * 16 + colL;
    float bias = b2[col];
    #pragma unroll
    for (int r = 0; r < 8; ++r)
      SCR[(rb + r) * 64 + col] = c[r] + bias;
  }
  __syncthreads();

  // log_softmax: per-row max + logsumexp
  if (lane < 16) {
    const float* row = SCR + lane * 64;
    float m = row[0];
    for (int c2 = 1; c2 < 64; ++c2) m = fmaxf(m, row[c2]);
    float s = 0.f;
    for (int c2 = 0; c2 < 64; ++c2) s += expf(row[c2] - m);
    RM[lane] = m + logf(s);
  }
  __syncthreads();

  // coalesced act output: [B,64]
  float* oa = out_act + rowBase * 64;
  for (int idx = lane; idx < 16 * 64; idx += 32)
    oa[idx] = SCR[idx] - RM[idx >> 6];

  __syncthreads();   // done reading SCR; reuse for value branch

  // GEMM3: h_v = relu(X[16x288] * WvT^T + bv1) -> SCR[16x64] f32
  for (int nt = 0; nt < 4; ++nt) {
    v8f c = {0.f, 0.f, 0.f, 0.f, 0.f, 0.f, 0.f, 0.f};
    #pragma unroll
    for (int kt = 0; kt < 9; ++kt)
      c = wmma_f16(ldsA(X, XS, kt * 32, lane), ldgB(WvT, 288, nt * 16, kt * 32, lane), c);
    int col = nt * 16 + colL;
    float bias = bv1[col];
    #pragma unroll
    for (int r = 0; r < 8; ++r)
      SCR[(rb + r) * 64 + col] = fmaxf(c[r] + bias, 0.f);
  }
  __syncthreads();

  // value head: tanh(h_v . wv2 + bv2)
  if (lane < 16) {
    const float* row = SCR + lane * 64;
    float s = bv2[0];
    for (int c2 = 0; c2 < 64; ++c2) s += row[c2] * wv2[c2];
    out_val[rowBase + lane] = tanhf(s);
  }
}

// ---------------- launch ----------------

extern "C" void kernel_launch(void* const* d_in, const int* in_sizes, int n_in,
                              void* d_out, int out_size, void* d_ws, size_t ws_size,
                              hipStream_t stream) {
  (void)n_in; (void)out_size; (void)ws_size;
  const float* state = (const float*)d_in[0];
  const float* myp   = (const float*)d_in[1];
  const float* hisp  = (const float*)d_in[2];
  const float* w1    = (const float*)d_in[3];
  const float* b1    = (const float*)d_in[4];
  const float* w2    = (const float*)d_in[5];
  const float* b2    = (const float*)d_in[6];
  const float* wv1   = (const float*)d_in[7];
  const float* bv1   = (const float*)d_in[8];
  const float* wv2   = (const float*)d_in[9];
  const float* bv2   = (const float*)d_in[10];

  const int B = in_sizes[0] / 128;             // 65536

  half_t* WcT = (half_t*)d_ws;                 // [288][128]
  half_t* W1T = WcT + 288 * 128;               // [128][288]
  half_t* W2T = W1T + 128 * 288;               // [64][128]
  half_t* WvT = W2T + 64 * 128;                // [64][288]

  float* out_act = (float*)d_out;              // [B,64]
  float* out_val = out_act + (size_t)B * 64;   // [B,1]

  net_prep<<<1, 256, 0, stream>>>(myp, hisp, w1, w2, wv1, WcT, W1T, W2T, WvT);

  const int rowsPerBlock = 16 * WAVES;         // 128
  const int blocks = B / rowsPerBlock;         // 512
  net_main<<<blocks, 256, WAVES * PER_WAVE, stream>>>(
      state, b1, b2, bv1, wv2, bv2, WcT, W1T, W2T, WvT, out_act, out_val);
}
